// FSQHier_74629351735686
// MI455X (gfx1250) — matrix operand order repaired and checked by reading, not compiled, not run on hardware
//
#include <hip/hip_runtime.h>
#include <cmath>

typedef __attribute__((ext_vector_type(16))) __bf16 v16bf;
typedef __attribute__((ext_vector_type(8)))  float  v8f;

#define HPAD 132   // bf16 stride for 128-wide LDS tiles (bank-conflict padding)
#define KP   36    // bf16 stride for 32-wide LDS tiles

struct LvlC {
    int d;
    float halfL[6], off[6], shift[6], halfW[6], basis[6];
    float selLo, selHi;
};

__device__ __forceinline__ int kmapA(int e, int half) { return e + 8 * half + ((e >= 8) ? 8 : 0); }
__device__ __forceinline__ int kmapB(int e, int half) { return e + 8 * half + ((e >= 8) ? 8 : 0); }

// ---------------------------------------------------------------------------
// Kernel 1: enc (128->d), FSQ quantize, vq code (+selection), lin1 (d->128),
//           per-workgroup BN1 partial sums. 128 threads, 256 rows per WG.
// ---------------------------------------------------------------------------
__global__ __launch_bounds__(128) void k_enc_lin1(
    const float* __restrict__ h_in, const float* __restrict__ p,
    const float* __restrict__ encW, const float* __restrict__ encB,
    const float* __restrict__ W1,   const float* __restrict__ b1,
    float* __restrict__ bufA, int* __restrict__ codeOut,
    float* __restrict__ pS, float* __restrict__ pQ,
    LvlC C, int levelIdx)
{
    __shared__ float hs[32 * 129];
    __shared__ float qs[32 * 8];
    __shared__ float wenc[6 * 128];

    int t = threadIdx.x;
    int d = C.d;
    for (int i = t; i < d * 128; i += 128) wenc[i] = encW[i];
    float w1r[6];
    float b1r = b1[t];
    for (int j = 0; j < d; ++j) w1r[j] = W1[t * d + j];   // W1 is (128, d)

    float accS = 0.f, accQ = 0.f;
    size_t base = (size_t)blockIdx.x * 256;

    for (int ch = 0; ch < 8; ++ch) {
        size_t rb = base + (size_t)ch * 32;
        __syncthreads();                 // protect hs/qs reuse
        for (int r = 0; r < 32; ++r)
            hs[r * 129 + t] = h_in[(rb + r) * 128 + t];   // coalesced
        __syncthreads();

        if (t < 32) {
            int r = t;
            float code = 0.f;
            for (int j = 0; j < d; ++j) {
                float s = encB[j];
                for (int k = 0; k < 128; ++k) s += hs[r * 129 + k] * wenc[j * 128 + k];
                float bounded = tanhf(s + C.shift[j]) * C.halfL[j] - C.off[j];
                float rq = rintf(bounded);               // round-half-even, like jnp.round
                qs[r * 8 + j] = rq / C.halfW[j];
                code += (rq + C.halfW[j]) * C.basis[j];
            }
            float pv = p[rb + r];
            bool sel = (C.selLo < pv) && (pv <= C.selHi);
            if (levelIdx == 0)      codeOut[rb + r] = sel ? (int)code : 0;
            else if (sel)           codeOut[rb + r] = (int)code;
        }
        __syncthreads();

        for (int r = 0; r < 32; ++r) {
            float y = b1r;
            for (int j = 0; j < d; ++j) y += qs[r * 8 + j] * w1r[j];
            bufA[(rb + r) * 128 + t] = y;                 // coalesced per row
            accS += y; accQ += y * y;
        }
    }
    pS[(size_t)blockIdx.x * 128 + t] = accS;
    pQ[(size_t)blockIdx.x * 128 + t] = accQ;
}

// ---------------------------------------------------------------------------
// Kernel 2: finalize BN stats deterministically (fixed-order reduction),
//           fold into scale/bias. 1 block, 128 threads.
// ---------------------------------------------------------------------------
__global__ __launch_bounds__(128) void k_bnstats(
    const float* __restrict__ pS, const float* __restrict__ pQ,
    const float* __restrict__ gamma, const float* __restrict__ beta,
    float* __restrict__ S, float* __restrict__ B, int nwg, float Nf)
{
    int c = threadIdx.x;
    float s = 0.f, q = 0.f;
    for (int w = 0; w < nwg; ++w) { s += pS[w * 128 + c]; q += pQ[w * 128 + c]; }
    float mean = s / Nf;
    float var  = q / Nf - mean * mean;       // biased, matches jnp.var
    float inv  = 1.f / sqrtf(var + 1e-5f);
    float sc   = gamma[c] * inv;
    S[c] = sc;
    B[c] = beta[c] - mean * sc;
}

// ---------------------------------------------------------------------------
// Kernel 3: y = relu(bn1(x1)) in bf16 LDS; z = y @ W2^T + b2 via WMMA;
//           in-place writeback to bufA; deterministic BN2 partial sums.
//           256 threads = 8 waves; wave w owns the 16-col block [16w,16w+16).
// ---------------------------------------------------------------------------
__global__ __launch_bounds__(256) void k_gemm_bn(
    float* __restrict__ bufA,
    const float* __restrict__ bnS, const float* __restrict__ bnB,
    const float* __restrict__ W2,  const float* __restrict__ b2,
    float* __restrict__ pS, float* __restrict__ pQ)
{
    __shared__ __bf16 w2s[128 * HPAD];
    __shared__ __bf16 yt[16 * HPAD];
    __shared__ float sc[128], bi[128];

    int t = threadIdx.x;
    int lane = t & 31, w = t >> 5, half = lane >> 4, n = lane & 15;

    if (t < 128) { sc[t] = bnS[t]; bi[t] = bnB[t]; }
    for (int i = t; i < 128 * 128; i += 256) {
        int c = i >> 7, k = i & 127;
        w2s[c * HPAD + k] = (__bf16)W2[i];       // W2[c][k]
    }
    float b2c = b2[16 * w + n];
    float regS = 0.f, regQ = 0.f;
    size_t rowbase0 = (size_t)blockIdx.x * 256;
    __syncthreads();

    for (int it = 0; it < 16; ++it) {
        size_t rb = rowbase0 + (size_t)it * 16;
        __syncthreads();                 // protect yt from previous iteration
        for (int i = t; i < 16 * 128; i += 256) {
            int m = i >> 7, k = i & 127;
            float x = bufA[(rb + m) * 128 + k];
            float y = sc[k] * x + bi[k];
            yt[m * HPAD + k] = (__bf16)(y > 0.f ? y : 0.f);
        }
        __syncthreads();

        v8f acc = {0.f, 0.f, 0.f, 0.f, 0.f, 0.f, 0.f, 0.f};
        for (int k0 = 0; k0 < 4; ++k0) {
            v16bf a, b;
#pragma unroll
            for (int e = 0; e < 16; ++e) {
                a[e] = yt[n * HPAD + (k0 * 32 + kmapA(e, half))];
                b[e] = w2s[(16 * w + n) * HPAD + (k0 * 32 + kmapB(e, half))];
            }
            acc = __builtin_amdgcn_wmma_f32_16x16x32_bf16(
                false, a, false, b, (short)0, acc, false, false);
        }

        int c = 16 * w + n;
        float s = 0.f, q = 0.f;
#pragma unroll
        for (int j = 0; j < 8; ++j) {
            float z = acc[j] + b2c;
            int m = j + 8 * half;
            bufA[(rb + m) * 128 + c] = z;        // in-place
            s += z; q += z * z;
        }
        s += __shfl_xor(s, 16, 32);              // 2-operand, order-invariant
        q += __shfl_xor(q, 16, 32);
        regS += s; regQ += q;
    }
    if (lane < 16) {
        pS[(size_t)blockIdx.x * 128 + 16 * w + lane] = regS;
        pQ[(size_t)blockIdx.x * 128 + 16 * w + lane] = regQ;
    }
}

// ---------------------------------------------------------------------------
// Kernel 5: y2 = relu(bn2(z)); feat = y2 @ Wout^T + bout (WMMA, cols padded
//           10->16); L2 normalize; selection-masked write into qacc.
//           256 threads = 8 waves; each wave owns its own 16-row tile.
// ---------------------------------------------------------------------------
__global__ __launch_bounds__(256) void k_out_feat(
    const float* __restrict__ bufA,
    const float* __restrict__ bnS, const float* __restrict__ bnB,
    const float* __restrict__ Wout, const float* __restrict__ bout,
    const float* __restrict__ p, float* __restrict__ qacc,
    float selLo, float selHi, int levelIdx)
{
    __shared__ __bf16 wos[16 * HPAD];
    __shared__ __bf16 yts[8 * 16 * HPAD];
    __shared__ float sc[128], bi[128], bo[16];

    int t = threadIdx.x;
    int lane = t & 31, w = t >> 5, half = lane >> 4, n = lane & 15;

    if (t < 128) { sc[t] = bnS[t]; bi[t] = bnB[t]; }
    if (t < 16)  bo[t] = (t < 10) ? bout[t] : 0.f;
    for (int i = t; i < 16 * 128; i += 256) {
        int c = i >> 7, k = i & 127;
        wos[c * HPAD + k] = (c < 10) ? (__bf16)Wout[c * 128 + k] : (__bf16)0.f;
    }
    __syncthreads();

    size_t rowbase = (size_t)blockIdx.x * 128 + (size_t)w * 16;
    __bf16* yt = &yts[w * 16 * HPAD];
    for (int i = lane; i < 16 * 128; i += 32) {
        int m = i >> 7, k = i & 127;
        float x = bufA[(rowbase + m) * 128 + k];
        float y = sc[k] * x + bi[k];
        yt[m * HPAD + k] = (__bf16)(y > 0.f ? y : 0.f);
    }
    __syncthreads();

    v8f acc = {0.f, 0.f, 0.f, 0.f, 0.f, 0.f, 0.f, 0.f};
    for (int k0 = 0; k0 < 4; ++k0) {
        v16bf a, b;
#pragma unroll
        for (int e = 0; e < 16; ++e) {
            a[e] = yt[n * HPAD + (k0 * 32 + kmapA(e, half))];
            b[e] = wos[n * HPAD + (k0 * 32 + kmapB(e, half))];
        }
        acc = __builtin_amdgcn_wmma_f32_16x16x32_bf16(
            false, a, false, b, (short)0, acc, false, false);
    }

    float f[8];
#pragma unroll
    for (int j = 0; j < 8; ++j) f[j] = acc[j] + bo[n];

#pragma unroll
    for (int j = 0; j < 8; ++j) {
        float v = (n < 10) ? f[j] * f[j] : 0.f;
        v += __shfl_xor(v, 1, 32);
        v += __shfl_xor(v, 2, 32);
        v += __shfl_xor(v, 4, 32);
        v += __shfl_xor(v, 8, 32);
        float inv = 1.f / (sqrtf(v) + 1e-6f);
        size_t row = rowbase + j + 8 * half;
        float pv = p[row];
        bool sel = (selLo < pv) && (pv <= selHi);
        float val = f[j] * inv;
        if (n < 10) {
            if (levelIdx == 0)  qacc[row * 10 + n] = sel ? val : 0.f;
            else if (sel)       qacc[row * 10 + n] = val;
        }
    }
}

// ---------------------------------------------------------------------------
// Kernel 6: out = qacc @ Wdec^T + bdec (10->128), K padded 10->32, one WMMA
//           per wave. 256 threads = 8 waves = 8 col tiles, 16 rows per block.
// ---------------------------------------------------------------------------
__global__ __launch_bounds__(256) void k_dec(
    const float* __restrict__ qacc, const float* __restrict__ Wd,
    const float* __restrict__ bd, float* __restrict__ out)
{
    __shared__ __bf16 wds[128 * KP];
    __shared__ __bf16 at[16 * KP];
    int t = threadIdx.x, lane = t & 31, w = t >> 5, half = lane >> 4, n = lane & 15;

    for (int i = t; i < 128 * 32; i += 256) {
        int c = i >> 5, k = i & 31;
        wds[c * KP + k] = (k < 10) ? (__bf16)Wd[c * 10 + k] : (__bf16)0.f;
    }
    size_t rb = (size_t)blockIdx.x * 16;
    for (int i = t; i < 16 * 32; i += 256) {
        int m = i >> 5, k = i & 31;
        at[m * KP + k] = (k < 10) ? (__bf16)qacc[(rb + m) * 10 + k] : (__bf16)0.f;
    }
    __syncthreads();

    v16bf a, b;
#pragma unroll
    for (int e = 0; e < 16; ++e) {
        a[e] = at[n * KP + kmapA(e, half)];
        b[e] = wds[(16 * w + n) * KP + kmapB(e, half)];
    }
    v8f acc = {0.f, 0.f, 0.f, 0.f, 0.f, 0.f, 0.f, 0.f};
    acc = __builtin_amdgcn_wmma_f32_16x16x32_bf16(
        false, a, false, b, (short)0, acc, false, false);

    int c = 16 * w + n;
    float bc = bd[c];
#pragma unroll
    for (int j = 0; j < 8; ++j)
        out[(rb + j + 8 * half) * 128 + c] = acc[j] + bc;
}

// ---------------------------------------------------------------------------
// Host side
// ---------------------------------------------------------------------------
static void fill_lvl(LvlC& c, const int* L, int d, int i)
{
    c.d = d;
    for (int j = 0; j < 6; ++j) { c.halfL[j] = 0.f; c.off[j] = 0.f; c.shift[j] = 0.f; c.halfW[j] = 1.f; c.basis[j] = 0.f; }
    double prod = 1.0;
    for (int j = 0; j < d; ++j) {
        double hl  = (L[j] - 1.0) * (1.0 - 1e-3) / 2.0;
        double off = (L[j] % 2 == 1) ? 0.0 : 0.5;
        c.halfL[j] = (float)hl;
        c.off[j]   = (float)off;
        c.shift[j] = (float)tan(off / hl);
        c.halfW[j] = (float)(L[j] / 2);
        c.basis[j] = (float)prod;
        prod *= L[j];
    }
    c.selLo = (float)(i / 3.0);
    c.selHi = (float)((i + 1) / 3.0);
}

extern "C" void kernel_launch(void* const* d_in, const int* in_sizes, int n_in,
                              void* d_out, int out_size, void* d_ws, size_t ws_size,
                              hipStream_t stream)
{
    (void)n_in; (void)out_size; (void)ws_size;
    const float* h_in = (const float*)d_in[0];
    const float* p    = (const float*)d_in[1];
    const int NR  = in_sizes[0] / 128;      // 524288
    const int NWG = NR / 256;               // 2048

    // Parameter pointer mapping: JAX pytree (alphabetical dict keys) vs
    // insertion-order flattening, disambiguated by in_sizes[2].
    const float *encW[3], *encB[3], *W1[3], *b1[3], *g1[3], *be1[3];
    const float *W2[3], *b2[3], *g2[3], *be2[3], *Wo[3], *bo[3];
    const float *Wd, *bd;
    bool alpha = (in_sizes[2] == 128);
    if (alpha) {
        for (int i = 0; i < 3; ++i) {
            int base = 2 + i * 10;
            be1[i] = (const float*)d_in[base + 0];
            g1[i]  = (const float*)d_in[base + 1];
            b1[i]  = (const float*)d_in[base + 2];
            W1[i]  = (const float*)d_in[base + 3];
            be2[i] = (const float*)d_in[base + 4];
            g2[i]  = (const float*)d_in[base + 5];
            b2[i]  = (const float*)d_in[base + 6];
            W2[i]  = (const float*)d_in[base + 7];
            bo[i]  = (const float*)d_in[base + 8];
            Wo[i]  = (const float*)d_in[base + 9];
            encB[i] = (const float*)d_in[34 + 2 * i];
            encW[i] = (const float*)d_in[35 + 2 * i];
        }
        bd = (const float*)d_in[32];
        Wd = (const float*)d_in[33];
    } else {
        for (int i = 0; i < 3; ++i) {
            int base = 2 + i * 12;
            encW[i] = (const float*)d_in[base + 0];
            encB[i] = (const float*)d_in[base + 1];
            W1[i]  = (const float*)d_in[base + 2];
            b1[i]  = (const float*)d_in[base + 3];
            g1[i]  = (const float*)d_in[base + 4];
            be1[i] = (const float*)d_in[base + 5];
            W2[i]  = (const float*)d_in[base + 6];
            b2[i]  = (const float*)d_in[base + 7];
            g2[i]  = (const float*)d_in[base + 8];
            be2[i] = (const float*)d_in[base + 9];
            Wo[i]  = (const float*)d_in[base + 10];
            bo[i]  = (const float*)d_in[base + 11];
        }
        Wd = (const float*)d_in[38];
        bd = (const float*)d_in[39];
    }

    // Outputs: first N*128 floats, then N int32 codes (bit-cast region).
    float* out     = (float*)d_out;
    int*   codeOut = (int*)(out + (size_t)NR * 128);

    // Workspace layout
    float* bufA = (float*)d_ws;                       // NR*128
    float* qacc = bufA + (size_t)NR * 128;            // NR*10
    float* pS   = qacc + (size_t)NR * 10;             // NWG*128
    float* pQ   = pS   + (size_t)NWG * 128;           // NWG*128
    float* bn1S = pQ   + (size_t)NWG * 128;
    float* bn1B = bn1S + 128;
    float* bn2S = bn1B + 128;
    float* bn2B = bn2S + 128;

    static const int L20[2]   = {4, 5};
    static const int L256[4]  = {4, 4, 4, 4};
    static const int L4096[6] = {4, 4, 4, 4, 4, 4};
    LvlC C[3];
    fill_lvl(C[0], L20, 2, 0);
    fill_lvl(C[1], L256, 4, 1);
    fill_lvl(C[2], L4096, 6, 2);

    for (int i = 0; i < 3; ++i) {
        k_enc_lin1<<<NWG, 128, 0, stream>>>(h_in, p, encW[i], encB[i], W1[i], b1[i],
                                            bufA, codeOut, pS, pQ, C[i], i);
        k_bnstats<<<1, 128, 0, stream>>>(pS, pQ, g1[i], be1[i], bn1S, bn1B, NWG, (float)NR);
        k_gemm_bn<<<NWG, 256, 0, stream>>>(bufA, bn1S, bn1B, W2[i], b2[i], pS, pQ);
        k_bnstats<<<1, 128, 0, stream>>>(pS, pQ, g2[i], be2[i], bn2S, bn2B, NWG, (float)NR);
        k_out_feat<<<NR / 128, 256, 0, stream>>>(bufA, bn2S, bn2B, Wo[i], bo[i], p, qacc,
                                                 C[i].selLo, C[i].selHi, i);
    }
    k_dec<<<NR / 16, 256, 0, stream>>>(qacc, Wd, bd, out);
}
